// WindowTransformerBlock_61220463837428
// MI455X (gfx1250) — compile-verified, matrix operand order checked by reading, not run on hardware
//
#include <hip/hip_runtime.h>

typedef __attribute__((ext_vector_type(16))) _Float16 v16h;
typedef __attribute__((ext_vector_type(8)))  _Float16 v8h;
typedef __attribute__((ext_vector_type(8)))  float    v8f;

#define NHEADS 12
#define HD 32
#define CDIM 384
#define TOK 65536          // 16 * 64 * 64 tokens
#define SHIFTV 4

// window-ordered token index -> normal (unrolled) token index (same mapping
// for the forward roll gather and the inverse roll scatter: +4 mod 64)
__device__ __forceinline__ int map_token(int wi) {
  int b    = wi >> 12;
  int rem  = wi & 4095;
  int widx = rem >> 6;
  int t    = rem & 63;
  int wy = widx >> 3, wx = widx & 7;
  int ty = t >> 3,    tx = t & 7;
  int y = (wy * 8 + ty + SHIFTV) & 63;
  int x = (wx * 8 + tx + SHIFTV) & 63;
  return (b << 12) + y * 64 + x;
}

__device__ __forceinline__ v8f wmma_f16(v16h a, v16h b, v8f c) {
  // (neg_a, A, neg_b, B, c_mod, C, reuse_a, reuse_b)
  return __builtin_amdgcn_wmma_f32_16x16x32_f16(false, a, false, b, (short)0, c,
                                                false, false);
}

// Build a v16h fragment from LDS: elements 0..7 at p, 8..15 at p+16 halves
// (matches ISA 16-bit A/B layout: VGPR0-3 = K base..+7, VGPR4-7 = K+16..+23)
__device__ __forceinline__ v16h ld_frag(const _Float16* p) {
  union { v16h v; v8h h[2]; } u;
  u.h[0] = *(const v8h*)(p);
  u.h[1] = *(const v8h*)(p + 16);
  return u.v;
}

// CDNA5 async copy: each active lane moves 16 bytes global -> LDS without
// touching VGPR data paths; tracked by ASYNCcnt.
__device__ __forceinline__ void async_ld_lds_b128(void* lds, const void* g) {
  uint32_t l = (uint32_t)(uintptr_t)lds;   // low 32 bits of generic ptr = LDS offset
  asm volatile("global_load_async_to_lds_b128 %0, %1, off"
               :: "v"(l), "v"(g) : "memory");
}
__device__ __forceinline__ void wait_asynccnt0() {
  asm volatile("s_wait_asynccnt 0" ::: "memory");
}

// ---------------------------------------------------------------- weights f32 -> f16
__global__ __launch_bounds__(256) void cvt_f16_kernel(const float* __restrict__ s,
                                                      _Float16* __restrict__ d, int n) {
  int i = blockIdx.x * 256 + threadIdx.x;
  if (i < n) d[i] = (_Float16)s[i];
}

// ---------------------------------------------------------------- LayerNorm (+optional window gather)
// one wave (32 lanes) per token; lane owns 12 strided channels
__global__ __launch_bounds__(256) void ln_kernel(const float* __restrict__ x,
                                                 const float* __restrict__ g,
                                                 const float* __restrict__ bta,
                                                 _Float16* __restrict__ out,
                                                 int do_map) {
  int wave = threadIdx.x >> 5;
  int lane = threadIdx.x & 31;
  int wi = blockIdx.x * 8 + wave;
  int src = do_map ? map_token(wi) : wi;
  const float* row = x + (size_t)src * CDIM;
  float v[12];
  float s = 0.f, s2 = 0.f;
#pragma unroll
  for (int j = 0; j < 12; j++) {
    float t = row[lane + j * 32];
    v[j] = t; s += t; s2 += t * t;
  }
#pragma unroll
  for (int m = 16; m > 0; m >>= 1) {
    s  += __shfl_xor(s,  m, 32);
    s2 += __shfl_xor(s2, m, 32);
  }
  float mu   = s  * (1.0f / CDIM);
  float var  = s2 * (1.0f / CDIM) - mu * mu;
  float rstd = rsqrtf(var + 1e-5f);
  _Float16* orow = out + (size_t)wi * CDIM;
#pragma unroll
  for (int j = 0; j < 12; j++) {
    int c = lane + j * 32;
    orow[c] = (_Float16)((v[j] - mu) * rstd * g[c] + bta[c]);
  }
}

// ---------------------------------------------------------------- tiled WMMA GEMM
// C[m,n] = sum_k A[m,k] * B[n,k] + bias[n]   (A:[M,K] f16, B:[N,K] f16 = weight rows)
// WG tile 128x128x32, 8 waves as 2(M) x 4(N), wave tile 64x32 (8 wmma / k-step).
// Double-buffered LDS fed by async global->LDS copies (ASYNCcnt).
// EPI: 0 = store f16
//      1 = proj: scatter to unwindowed pos, + shortcut, store f32
//      2 = GELU(exact), store f16
//      3 = + residual, store f32 (final output)
template <int EPI>
__global__ __launch_bounds__(256) void gemm_wmma_kernel(
    const _Float16* __restrict__ A, const _Float16* __restrict__ B,
    const float* __restrict__ bias, void* __restrict__ outp,
    const float* __restrict__ resid, int M, int N, int K) {
  __shared__ __align__(16) _Float16 a_s[2][128][40];   // 32 data + 8 pad (80B pitch)
  __shared__ __align__(16) _Float16 b_s[2][128][40];
  const int tid  = threadIdx.x;
  const int wave = tid >> 5, lane = tid & 31;
  const int half = lane >> 4, lr = lane & 15;
  const int wm = wave & 1, wn = wave >> 1;
  const int m0 = blockIdx.x * 128, n0 = blockIdx.y * 128;
  const int koff = half * 8;
  const int ld_row = tid >> 2;            // 0..63
  const int ld_ch  = (tid & 3) * 8;       // half-element offset within 32-wide k slab

  v8f zero = {};
  v8f acc[4][2];
#pragma unroll
  for (int i = 0; i < 4; i++)
#pragma unroll
    for (int j = 0; j < 2; j++) acc[i][j] = zero;

  auto issue_tile = [&](int kt, int bufi) {
#pragma unroll
    for (int i = 0; i < 2; i++) {
      int row = ld_row + i * 64;
      async_ld_lds_b128(&a_s[bufi][row][ld_ch],
                        A + (size_t)(m0 + row) * K + kt + ld_ch);
      async_ld_lds_b128(&b_s[bufi][row][ld_ch],
                        B + (size_t)(n0 + row) * K + kt + ld_ch);
    }
  };

  const int nk = K >> 5;
  issue_tile(0, 0);
  wait_asynccnt0();
  __syncthreads();

  for (int kt = 0; kt < nk; kt++) {
    const int cur = kt & 1;
    if (kt + 1 < nk) issue_tile((kt + 1) << 5, cur ^ 1);  // prefetch next slab

    v16h bf[2];
#pragma unroll
    for (int nt = 0; nt < 2; nt++)
      bf[nt] = ld_frag(&b_s[cur][wn * 32 + nt * 16 + lr][koff]);
#pragma unroll
    for (int mt = 0; mt < 4; mt++) {
      v16h af = ld_frag(&a_s[cur][wm * 64 + mt * 16 + lr][koff]);
#pragma unroll
      for (int nt = 0; nt < 2; nt++) acc[mt][nt] = wmma_f16(af, bf[nt], acc[mt][nt]);
    }
    wait_asynccnt0();   // next buffer landed (overlapped with compute above)
    __syncthreads();    // everyone done reading cur + writing next
  }

#pragma unroll
  for (int mt = 0; mt < 4; mt++)
#pragma unroll
    for (int nt = 0; nt < 2; nt++) {
      int n = n0 + wn * 32 + nt * 16 + lr;
      float bv = bias[n];
#pragma unroll
      for (int r = 0; r < 8; r++) {
        int m = m0 + wm * 64 + mt * 16 + half * 8 + r;
        float val = acc[mt][nt][r] + bv;
        if constexpr (EPI == 0) {
          ((_Float16*)outp)[(size_t)m * N + n] = (_Float16)val;
        } else if constexpr (EPI == 1) {
          int ni = map_token(m);
          size_t o = (size_t)ni * N + n;
          ((float*)outp)[o] = val + resid[o];
        } else if constexpr (EPI == 2) {
          float gl = 0.5f * val * (1.0f + erff(val * 0.70710678118654752f));
          ((_Float16*)outp)[(size_t)m * N + n] = (_Float16)gl;
        } else {
          size_t o = (size_t)m * N + n;
          ((float*)outp)[o] = val + resid[o];
        }
      }
    }
}

// ---------------------------------------------------------------- window attention
// one block = one (window, head): scores = q k^T / sqrt(HD) + bias; softmax; o = attn v
__global__ __launch_bounds__(128) void attn_kernel(const _Float16* __restrict__ qkv,
                                                   const float* __restrict__ rpb,
                                                   _Float16* __restrict__ obuf) {
  __shared__ __align__(16) _Float16 q_s[64][40];
  __shared__ __align__(16) _Float16 k_s[64][40];
  __shared__ __align__(16) _Float16 vT_s[32][72];   // v transposed: [d][token]
  __shared__ __align__(16) _Float16 at_s[64][72];   // softmaxed attn (f16)
  __shared__ float sc_s[64][65];                    // scores (f32)
  const int win = blockIdx.x, head = blockIdx.y;
  const int tid = threadIdx.x;
  const size_t base = (size_t)win * 64 * 1152;
  const int hoff = head * HD;

  // q,k: async 16B-chunk copies (64 rows x 4 chunks each)
#pragma unroll
  for (int i = 0; i < 2; i++) {
    int id = tid + i * 128;                // 0..255
    int t = id >> 2, ch = (id & 3) * 8;
    size_t tb = base + (size_t)t * 1152 + hoff + ch;
    async_ld_lds_b128(&q_s[t][ch], qkv + tb);
    async_ld_lds_b128(&k_s[t][ch], qkv + tb + 384);
  }
  // v: transposed stage (element scatter, manual)
  for (int i = tid; i < 2048; i += 128) {
    int t = i >> 5, d = i & 31;
    vT_s[d][t] = qkv[base + (size_t)t * 1152 + hoff + 768 + d];
  }
  wait_asynccnt0();
  __syncthreads();

  const int wave = tid >> 5, lane = tid & 31;
  const int half = lane >> 4, lr = lane & 15;
  const int koff = half * 8;

  // scores: wave owns query rows [wave*16, wave*16+16)
  {
    v16h af = ld_frag(&q_s[wave * 16 + lr][koff]);
#pragma unroll
    for (int nt = 0; nt < 4; nt++) {
      v16h bf = ld_frag(&k_s[nt * 16 + lr][koff]);
      v8f c = {};
      c = wmma_f16(af, bf, c);
#pragma unroll
      for (int r = 0; r < 8; r++)
        sc_s[wave * 16 + half * 8 + r][nt * 16 + lr] = c[r] * 0.17677669529663687f;
    }
  }
  __syncthreads();

  // bias + softmax: one thread per query row (threads 0..63)
  if (tid < 64) {
    int i = tid, ty = i >> 3, tx = i & 7;
    float mx = -1e30f;
#pragma unroll 8
    for (int j = 0; j < 64; j++) {
      int jy = j >> 3, jx = j & 7;
      int ridx = (ty - jy + 7) * 15 + (tx - jx + 7);
      float s = sc_s[i][j] + rpb[ridx * NHEADS + head];
      sc_s[i][j] = s;
      mx = fmaxf(mx, s);
    }
    float sum = 0.f;
#pragma unroll 8
    for (int j = 0; j < 64; j++) {
      float e = __expf(sc_s[i][j] - mx);
      sc_s[i][j] = e;
      sum += e;
    }
    float inv = 1.0f / sum;
#pragma unroll 8
    for (int j = 0; j < 64; j++) at_s[i][j] = (_Float16)(sc_s[i][j] * inv);
  }
  __syncthreads();

  // o = attn @ v : M=64(4 waves), N=32(2 tiles), K=64(2 steps)
  {
    v8f c[2] = {{}, {}};
#pragma unroll
    for (int ks = 0; ks < 2; ks++) {
      v16h af = ld_frag(&at_s[wave * 16 + lr][ks * 32 + koff]);
#pragma unroll
      for (int nt = 0; nt < 2; nt++) {
        v16h bf = ld_frag(&vT_s[nt * 16 + lr][ks * 32 + koff]);
        c[nt] = wmma_f16(af, bf, c[nt]);
      }
    }
#pragma unroll
    for (int nt = 0; nt < 2; nt++)
#pragma unroll
      for (int r = 0; r < 8; r++) {
        int t = wave * 16 + half * 8 + r;
        int d = nt * 16 + lr;
        obuf[((size_t)win * 64 + t) * CDIM + hoff + d] = (_Float16)c[nt][r];
      }
  }
}

// ---------------------------------------------------------------- launcher
extern "C" void kernel_launch(void* const* d_in, const int* in_sizes, int n_in,
                              void* d_out, int out_size, void* d_ws, size_t ws_size,
                              hipStream_t stream) {
  (void)in_sizes; (void)n_in; (void)out_size; (void)ws_size;
  const float* x      = (const float*)d_in[0];
  const float* n1g    = (const float*)d_in[1];
  const float* n1b    = (const float*)d_in[2];
  const float* qkv_w  = (const float*)d_in[3];
  const float* qkv_b  = (const float*)d_in[4];
  const float* proj_w = (const float*)d_in[5];
  const float* proj_b = (const float*)d_in[6];
  const float* rpb    = (const float*)d_in[7];
  const float* n2g    = (const float*)d_in[8];
  const float* n2b    = (const float*)d_in[9];
  const float* fc1_w  = (const float*)d_in[10];
  const float* fc1_b  = (const float*)d_in[11];
  const float* fc2_w  = (const float*)d_in[12];
  const float* fc2_b  = (const float*)d_in[13];
  float* out = (float*)d_out;

  char* ws = (char*)d_ws;
  _Float16* h_win = (_Float16*)ws; ws += (size_t)TOK * 384 * 2;   // LN1, window order
  _Float16* qkvb  = (_Float16*)ws; ws += (size_t)TOK * 1152 * 2;  // qkv (window order)
  _Float16* obuf  = (_Float16*)ws; ws += (size_t)TOK * 384 * 2;   // attn out (window order)
  float*    x1    = (float*)ws;    ws += (size_t)TOK * 384 * 4;   // shortcut + proj (normal order)
  _Float16* h2    = (_Float16*)ws; ws += (size_t)TOK * 384 * 2;   // LN2
  _Float16* m1    = (_Float16*)ws; ws += (size_t)TOK * 1536 * 2;  // GELU(fc1)
  _Float16* wq = (_Float16*)ws; ws += (size_t)1152 * 384 * 2;
  _Float16* wp = (_Float16*)ws; ws += (size_t)384 * 384 * 2;
  _Float16* w1 = (_Float16*)ws; ws += (size_t)1536 * 384 * 2;
  _Float16* w2 = (_Float16*)ws; ws += (size_t)384 * 1536 * 2;

  cvt_f16_kernel<<<(1152 * 384 + 255) / 256, 256, 0, stream>>>(qkv_w, wq, 1152 * 384);
  cvt_f16_kernel<<<(384 * 384 + 255) / 256, 256, 0, stream>>>(proj_w, wp, 384 * 384);
  cvt_f16_kernel<<<(1536 * 384 + 255) / 256, 256, 0, stream>>>(fc1_w, w1, 1536 * 384);
  cvt_f16_kernel<<<(384 * 1536 + 255) / 256, 256, 0, stream>>>(fc2_w, w2, 384 * 1536);

  // LN1 + roll + window partition
  ln_kernel<<<TOK / 8, 256, 0, stream>>>(x, n1g, n1b, h_win, 1);
  // qkv = h @ qkv_w^T + b        [65536 x 1152], K=384
  gemm_wmma_kernel<0><<<dim3(TOK / 128, 1152 / 128), 256, 0, stream>>>(
      h_win, wq, qkv_b, qkvb, nullptr, TOK, 1152, 384);
  // window attention (1024 windows x 12 heads)
  attn_kernel<<<dim3(1024, NHEADS), 128, 0, stream>>>(qkvb, rpb, obuf);
  // proj + inverse-roll scatter + shortcut -> x1  [65536 x 384], K=384
  gemm_wmma_kernel<1><<<dim3(TOK / 128, 384 / 128), 256, 0, stream>>>(
      obuf, wp, proj_b, x1, x, TOK, 384, 384);
  // LN2
  ln_kernel<<<TOK / 8, 256, 0, stream>>>(x1, n2g, n2b, h2, 0);
  // fc1 + GELU                   [65536 x 1536], K=384
  gemm_wmma_kernel<2><<<dim3(TOK / 128, 1536 / 128), 256, 0, stream>>>(
      h2, w1, fc1_b, m1, nullptr, TOK, 1536, 384);
  // fc2 + residual -> out        [65536 x 384], K=1536
  gemm_wmma_kernel<3><<<dim3(TOK / 128, 384 / 128), 256, 0, stream>>>(
      m1, w2, fc2_b, out, x1, TOK, 384, 1536);
}
